// NIC_Decoder_41369124995383
// MI455X (gfx1250) — matrix operand order repaired and checked by reading, not compiled, not run on hardware
//
#include <hip/hip_runtime.h>

// ---------------- problem dims ----------------
#define B_     64
#define T_     96
#define EMB_   512
#define UNITS_ 512
#define VOCAB_ 8000
#define FOURU_ 2048
#define MROWS_ (B_*T_)          // 6144
#define NWG_LSTM 16

// ---------------- CDNA5 types ----------------
typedef __attribute__((ext_vector_type(16))) __bf16       v16bf;
typedef __attribute__((ext_vector_type(8)))  float        v8f;
typedef __attribute__((ext_vector_type(4)))  unsigned int uintx4;
typedef __attribute__((ext_vector_type(4)))  int          i32x4;

union FragU { v16bf v; uintx4 q[2]; };

// ---------------- async global->LDS (gfx1250) ----------------
#if defined(__has_builtin)
#if __has_builtin(__builtin_amdgcn_global_load_async_to_lds_b128)
#define CDNA5_ASYNC_LDS 1
#endif
#endif

#if defined(CDNA5_ASYNC_LDS)
#if __has_builtin(__builtin_amdgcn_s_wait_asynccnt)
#define WAIT_ASYNC() __builtin_amdgcn_s_wait_asynccnt(0)
#else
#define WAIT_ASYNC() asm volatile("s_wait_asynccnt 0" ::: "memory")
#endif
#else
#define WAIT_ASYNC() ((void)0)
#endif

#if defined(CDNA5_ASYNC_LDS)
typedef __attribute__((address_space(1))) i32x4 g_i32x4;   // global (HIP prints as __device__)
typedef __attribute__((address_space(3))) i32x4 s_i32x4;   // LDS
#endif

__device__ __forceinline__ void copy16_to_lds(unsigned short* lds, const unsigned short* g) {
#if defined(CDNA5_ASYNC_LDS)
  __builtin_amdgcn_global_load_async_to_lds_b128(
      (g_i32x4*)g,      // addrspace(1) int4*  (src, per-lane global address)
      (s_i32x4*)lds,    // addrspace(3) int4*  (dst, per-lane LDS address)
      0, 0);
#else
  *(uintx4*)lds = *(const uintx4*)g;
#endif
}

__device__ __forceinline__ void cluster_barrier_hw() {
#if defined(__has_builtin)
#if __has_builtin(__builtin_amdgcn_s_cluster_barrier)
  __builtin_amdgcn_s_cluster_barrier();
#else
  asm volatile("s_barrier_signal -3\n\ts_barrier_wait -3" ::: "memory");
#endif
#else
  asm volatile("s_barrier_signal -3\n\ts_barrier_wait -3" ::: "memory");
#endif
}

// ---------------- small helpers ----------------
__device__ __forceinline__ unsigned short f32_to_bf16(float x) {
  unsigned u = __float_as_uint(x);
  unsigned r = u + 0x7FFFu + ((u >> 16) & 1u);   // round-to-nearest-even
  return (unsigned short)(r >> 16);
}

__device__ __forceinline__ unsigned ihash(unsigned x) {
  x ^= x >> 16; x *= 0x7feb352du;
  x ^= x >> 15; x *= 0x846ca68bu;
  x ^= x >> 16; return x;
}
// Bernoulli(keep=0.5) scaled by 1/keep=2.0 (stand-in for jax.random threefry)
__device__ __forceinline__ float bmask(unsigned idx, unsigned salt) {
  return (ihash(idx * 0x9e3779b9u + salt) & 1u) ? 2.0f : 0.0f;
}
__device__ __forceinline__ float sigm(float x) { return 1.0f / (1.0f + __expf(-x)); }

__device__ __forceinline__ v8f zero8() {
  v8f z;
#pragma unroll
  for (int i = 0; i < 8; ++i) z[i] = 0.f;
  return z;
}

// A-matrix 16x32 bf16 fragment (per ISA 7.12.2): lane half hi selects K {0..7,16..23} vs {8..15,24..31}
__device__ __forceinline__ v16bf load_fragA(const unsigned short* row, int k0 /* ks*32 + hi*8 */) {
  FragU f;
  f.q[0] = *(const uintx4*)(row + k0);
  f.q[1] = *(const uintx4*)(row + k0 + 16);
  return f.v;
}
// B-matrix 32x16 bf16 fragment: lane half hi selects K {0..7,16..23} vs {8..15,24..31}
__device__ __forceinline__ v16bf load_fragB(const unsigned short* row, int k0 /* ks*32 + hi*8 */) {
  FragU f;
  f.q[0] = *(const uintx4*)(row + k0);
  f.q[1] = *(const uintx4*)(row + k0 + 16);
  return f.v;
}

// ---------------- prep kernels ----------------
// dst (N x K) bf16 = transpose(src (K x N) f32); K == 512 here
__global__ void k_transpose_bf16(const float* __restrict__ src, unsigned short* __restrict__ dst,
                                 int K, int N) {
  int idx = blockIdx.x * blockDim.x + threadIdx.x;
  if (idx >= K * N) return;
  int n = idx / K, k = idx - n * K;
  dst[idx] = f32_to_bf16(src[(size_t)k * N + n]);
}

// x[b,0,:]=images_emb[b]; x[b,t,:]=E[targets[b,t-1]]*word_mask(b,e)  -> bf16 (6144 x 512)
__global__ void k_build_x(const float* __restrict__ img, const int* __restrict__ tgt,
                          const float* __restrict__ E, unsigned short* __restrict__ xbf) {
  int idx = blockIdx.x * blockDim.x + threadIdx.x;
  if (idx >= MROWS_ * EMB_) return;
  int e = idx & (EMB_ - 1);
  int r = idx >> 9;
  int b = r / T_, t = r - b * T_;
  float v;
  if (t == 0) {
    v = img[b * EMB_ + e];
  } else {
    int w = tgt[b * T_ + (t - 1)];
    v = E[(size_t)w * EMB_ + e] * bmask((unsigned)(b * EMB_ + e), 0x057D57D1u);
  }
  xbf[idx] = f32_to_bf16(v);
}

// ---------------- generic bf16 WMMA GEMM: C(MxN,f32) = A(MxK,bf16) @ Bt(NxK,bf16)^T + bias ----------
// tile: 128(M) x 64(N) per WG, 8 waves as 4x2, each wave 2x2 16x16 tiles, K staged 64 at a time
__global__ __launch_bounds__(256) void k_gemm_bf16(
    const unsigned short* __restrict__ A,
    const unsigned short* __restrict__ Bt,
    const float* __restrict__ bias,
    float* __restrict__ C,
    int M, int N, int K) {
  __shared__ unsigned short As[2][128 * 64];
  __shared__ unsigned short Bs[2][64 * 64];
  const int tid  = threadIdx.x;
  const int wave = tid >> 5, lane = tid & 31;
  const int lm = lane & 15, hi = lane >> 4;
  const int mbase = blockIdx.y * 128;
  const int nbase = blockIdx.x * 64;
  const int wm = (wave >> 1) * 32;
  const int wn = (wave & 1) * 32;

  auto stage = [&](int buf, int kb) {
#pragma unroll
    for (int i = 0; i < 4; ++i) {               // A: 128 rows * 128B = 1024 chunks
      int c = tid + i * 256;
      int row = c >> 3, seg = c & 7;
      copy16_to_lds(&As[buf][row * 64 + seg * 8],
                    A + (size_t)(mbase + row) * K + kb + seg * 8);
    }
#pragma unroll
    for (int i = 0; i < 2; ++i) {               // B: 64 rows * 128B = 512 chunks
      int c = tid + i * 256;
      int row = c >> 3, seg = c & 7;
      copy16_to_lds(&Bs[buf][row * 64 + seg * 8],
                    Bt + (size_t)(nbase + row) * K + kb + seg * 8);
    }
  };

  v8f acc[2][2];
#pragma unroll
  for (int i = 0; i < 2; ++i)
#pragma unroll
    for (int j = 0; j < 2; ++j) acc[i][j] = zero8();

  stage(0, 0);
  WAIT_ASYNC();
  __syncthreads();

  const int nkb = K >> 6;
  for (int kb = 0; kb < nkb; ++kb) {
    const int buf = kb & 1;
    if (kb + 1 < nkb) stage(buf ^ 1, (kb + 1) << 6);
#pragma unroll
    for (int ks = 0; ks < 2; ++ks) {
      const int ka = ks * 32 + hi * 8;
      const int kbf = ks * 32 + hi * 8;
      v16bf a0 = load_fragA(&As[buf][(wm + lm) * 64], ka);
      v16bf a1 = load_fragA(&As[buf][(wm + 16 + lm) * 64], ka);
      v16bf b0 = load_fragB(&Bs[buf][(wn + lm) * 64], kbf);
      v16bf b1 = load_fragB(&Bs[buf][(wn + 16 + lm) * 64], kbf);
      acc[0][0] = __builtin_amdgcn_wmma_f32_16x16x32_bf16(false, a0, false, b0, (short)0, acc[0][0], false, false);
      acc[0][1] = __builtin_amdgcn_wmma_f32_16x16x32_bf16(false, a0, false, b1, (short)0, acc[0][1], false, false);
      acc[1][0] = __builtin_amdgcn_wmma_f32_16x16x32_bf16(false, a1, false, b0, (short)0, acc[1][0], false, false);
      acc[1][1] = __builtin_amdgcn_wmma_f32_16x16x32_bf16(false, a1, false, b1, (short)0, acc[1][1], false, false);
    }
    WAIT_ASYNC();
    __syncthreads();
  }

#pragma unroll
  for (int mt = 0; mt < 2; ++mt)
#pragma unroll
    for (int nt = 0; nt < 2; ++nt)
#pragma unroll
      for (int r = 0; r < 8; ++r) {
        int row = mbase + wm + mt * 16 + r + 8 * hi;  // C/D layout: m = r + 8*hi
        int col = nbase + wn + nt * 16 + lm;          //             n = lane&15
        C[(size_t)row * N + col] = acc[mt][nt][r] + bias[col];
      }
}

// ---------------- persistent LSTM kernel (16 WGs, grid barrier per step) ----------------
__device__ __forceinline__ void grid_barrier(unsigned* cnt, unsigned gen) {
  __syncthreads();
  if (threadIdx.x == 0) {
    __hip_atomic_fetch_add(cnt, 1u, __ATOMIC_RELEASE, __HIP_MEMORY_SCOPE_AGENT);
    while (__hip_atomic_load(cnt, __ATOMIC_ACQUIRE, __HIP_MEMORY_SCOPE_AGENT) < gen * NWG_LSTM) { }
  }
  __syncthreads();
}

// WG g owns hidden columns [g*32, g*32+32) across all 4 gates.
// hm: double-buffered masked-h, [2][4][64][512] bf16.  rnn: (B,T,512) bf16 (final_mask applied).
__global__ __launch_bounds__(256) void k_lstm(
    const float* __restrict__ z,            // (B,T,2048) fp32, row = b*T+t
    const unsigned short* __restrict__ Ut,  // (2048 x 512) bf16, transposed U
    unsigned short* __restrict__ hm,
    unsigned short* __restrict__ rnn,
    unsigned* __restrict__ cnt) {
  __shared__ float cS[B_ * 32];        // cell state for owned columns
  __shared__ float preS[4 * B_ * 32];  // gate preactivations

  const int tid = threadIdx.x;
  const int wave = tid >> 5, lane = tid & 31;
  const int lm = lane & 15, hi = lane >> 4;
  const int u0 = blockIdx.x * 32;
  const int gate = wave >> 1, nt = wave & 1;   // 4 gates x 2 column halves = 8 waves

  for (int i = tid; i < B_ * 32; i += 256) cS[i] = 0.f;
  for (int i = tid; i < 4 * B_ * 32; i += 256) {   // h0 = 0 -> masked h buffer 0 = 0
    int k = i >> 11, b = (i >> 5) & 63, j = i & 31;
    hm[((k * B_ + b) << 9) + u0 + j] = 0;
  }
  unsigned bar = 0;
  grid_barrier(cnt, ++bar);

  // B fragment row: Ut row n = gate*512 + u0 + nt*16 + lm, K = u (512)
  const unsigned short* browp = Ut + (size_t)(gate * UNITS_ + u0 + nt * 16 + lm) * UNITS_;

  for (int t = 0; t < T_; ++t) {
    const unsigned short* hmr = hm + (size_t)(t & 1) * (4 * B_ * UNITS_) + (size_t)gate * B_ * UNITS_;
    unsigned short* hmw = hm + (size_t)((t + 1) & 1) * (4 * B_ * UNITS_);

    // pre(64x32) = hm_gate(64x512) @ U_gate[:, owned 32 cols]; wave does 4 Mtiles x its 16 cols
    v8f acc[4];
#pragma unroll
    for (int m = 0; m < 4; ++m) acc[m] = zero8();
#pragma unroll 4
    for (int kk = 0; kk < UNITS_ / 32; ++kk) {
      v16bf bf = load_fragB(browp, kk * 32 + hi * 8);
#pragma unroll
      for (int mt = 0; mt < 4; ++mt) {
        v16bf af = load_fragA(hmr + ((mt * 16 + lm) << 9), kk * 32 + hi * 8);
        acc[mt] = __builtin_amdgcn_wmma_f32_16x16x32_bf16(false, af, false, bf, (short)0, acc[mt], false, false);
      }
    }
#pragma unroll
    for (int mt = 0; mt < 4; ++mt)
#pragma unroll
      for (int r = 0; r < 8; ++r) {
        int b = mt * 16 + r + 8 * hi;
        int j = nt * 16 + lm;
        float pre = acc[mt][r] +
                    z[((size_t)(b * T_ + t) << 11) + gate * UNITS_ + u0 + j];
        preS[(gate * B_ + b) * 32 + j] = pre;
      }
    __syncthreads();

    for (int i = tid; i < B_ * 32; i += 256) {
      int b = i >> 5, j = i & 31;
      float iv = sigm(preS[(0 * B_ + b) * 32 + j]);
      float fv = sigm(preS[(1 * B_ + b) * 32 + j]);
      float gv = tanhf(preS[(2 * B_ + b) * 32 + j]);
      float ov = sigm(preS[(3 * B_ + b) * 32 + j]);
      float c2 = fv * cS[i] + iv * gv;
      float h2 = ov * tanhf(c2);
      cS[i] = c2;
      int u = u0 + j;
#pragma unroll
      for (int k = 0; k < 4; ++k)   // four rec_mask'd copies for next step's gate GEMMs
        hmw[((k * B_ + b) << 9) + u] =
            f32_to_bf16(h2 * bmask((unsigned)((k * B_ + b) * UNITS_ + u), 0xA5A50001u));
      rnn[((size_t)(b * T_ + t) << 9) + u] =
          f32_to_bf16(h2 * bmask((unsigned)((b * T_ + t) * UNITS_ + u), 0xC3C30001u));
    }
    cluster_barrier_hw();        // real barrier if launched as a 16-WG cluster; NOP otherwise
    grid_barrier(cnt, ++bar);    // portable device-scope barrier
  }
}

// ---------------- host entry ----------------
extern "C" void kernel_launch(void* const* d_in, const int* in_sizes, int n_in,
                              void* d_out, int out_size, void* d_ws, size_t ws_size,
                              hipStream_t stream) {
  (void)in_sizes; (void)n_in; (void)out_size; (void)ws_size;
  const float* img  = (const float*)d_in[1];   // d_in[0] = pos_embs (unused by reference)
  const int*   tgt  = (const int*)d_in[2];
  const float* E    = (const float*)d_in[3];
  const float* W    = (const float*)d_in[4];
  const float* U    = (const float*)d_in[5];
  const float* bb   = (const float*)d_in[6];
  const float* Wout = (const float*)d_in[7];
  const float* bout = (const float*)d_in[8];
  float* out = (float*)d_out;

  char* ws = (char*)d_ws;
  size_t off = 0;
  auto salloc = [&](size_t bytes) -> char* {
    char* p = ws + off;
    off = (off + bytes + 255) & ~(size_t)255;
    return p;
  };
  unsigned short* xbf   = (unsigned short*)salloc((size_t)MROWS_ * EMB_ * 2);
  unsigned short* Wt    = (unsigned short*)salloc((size_t)FOURU_ * EMB_ * 2);
  unsigned short* Utp   = (unsigned short*)salloc((size_t)FOURU_ * UNITS_ * 2);
  unsigned short* Woutt = (unsigned short*)salloc((size_t)VOCAB_ * UNITS_ * 2);
  float*          zbuf  = (float*)salloc((size_t)MROWS_ * FOURU_ * 4);
  unsigned short* hm    = (unsigned short*)salloc((size_t)2 * 4 * B_ * UNITS_ * 2);
  unsigned short* rnn   = (unsigned short*)salloc((size_t)MROWS_ * UNITS_ * 2);
  unsigned*       cnt   = (unsigned*)salloc(256);

  (void)hipMemsetAsync(cnt, 0, 256, stream);

  { int n = EMB_ * FOURU_;  k_transpose_bf16<<<(n + 255) / 256, 256, 0, stream>>>(W, Wt, EMB_, FOURU_); }
  { int n = UNITS_ * FOURU_; k_transpose_bf16<<<(n + 255) / 256, 256, 0, stream>>>(U, Utp, UNITS_, FOURU_); }
  { int n = UNITS_ * VOCAB_; k_transpose_bf16<<<(n + 255) / 256, 256, 0, stream>>>(Wout, Woutt, UNITS_, VOCAB_); }
  { int n = MROWS_ * EMB_;  k_build_x<<<(n + 255) / 256, 256, 0, stream>>>(img, tgt, E, xbf); }

  { dim3 g(FOURU_ / 64, MROWS_ / 128);
    k_gemm_bf16<<<g, 256, 0, stream>>>(xbf, Wt, bb, zbuf, MROWS_, FOURU_, EMB_); }

  k_lstm<<<NWG_LSTM, 256, 0, stream>>>(zbuf, Utp, hm, rnn, cnt);

  { dim3 g(VOCAB_ / 64, MROWS_ / 128);
    k_gemm_bf16<<<g, 256, 0, stream>>>(rnn, Woutt, bout, out, MROWS_, VOCAB_, UNITS_); }
}
